// QuasarMoE_50182397886794
// MI455X (gfx1250) — matrix operand order repaired and checked by reading, not compiled, not run on hardware
//
#include <hip/hip_runtime.h>
#include <hip/hip_bf16.h>
#include <math.h>

// Problem dims (compile-time, from reference): B=2 S=512 H=512 I=1024 E=64 K=4
#define Hd 512
#define Id 1024
#define Ed 64
#define Kd 4
#define Td 1024  // B*S tokens
#define MB 32    // tokens per workgroup block (two 16-row WMMA tiles)

typedef __attribute__((ext_vector_type(16))) __bf16 v16bf;
typedef __attribute__((ext_vector_type(8)))  __bf16 v8bf;
typedef __attribute__((ext_vector_type(8)))  float  v8f;

union BFrag { v16bf v; v8bf h[2]; __bf16 e[16]; };

// ---- WMMA fragment loaders ------------------------------------------------
// A (16x32 bf16, M x K): lane m = lane%16, half = lane/16.
// elements 0..15 -> k = kb + 16*half + 0..15   (contiguous in row-major LDS tile)
__device__ __forceinline__ v16bf load_a_lds(const __bf16* lds, int ld, int m,
                                            int kb, int half) {
  const __bf16* p = lds + m * ld + kb + 16 * half;
  BFrag a;
  a.h[0] = *(const v8bf*)(p);
  a.h[1] = *(const v8bf*)(p + 8);
  return a.v;
}

// B (32x16 bf16, K x N): lane n = lane%16, half = lane/16.
// elements 0..15 -> k = kb + 16*half + 0..15 of weight row n (row-major [N][Kdim]).
// Weights live in HBM as fp32 -> load 4x float4 (global_load_b128) and convert.
__device__ __forceinline__ v16bf load_b_f32(const float* __restrict__ wrow,
                                            int kb, int half) {
  const float* p = wrow + kb + 16 * half;
  float f[16];
#pragma unroll
  for (int j = 0; j < 4; ++j) {
    float4 v = ((const float4*)p)[j];
    f[4 * j + 0] = v.x; f[4 * j + 1] = v.y;
    f[4 * j + 2] = v.z; f[4 * j + 3] = v.w;
  }
  // prefetch the next k-chunk of this row (hides HBM/L2 latency of weight stream)
  __builtin_prefetch(p + 32, 0, 1);
  BFrag b;
#pragma unroll
  for (int j = 0; j < 16; ++j) b.e[j] = (__bf16)f[j];
  return b.v;
}

#define WMMA_BF16(A, B, C) \
  __builtin_amdgcn_wmma_f32_16x16x32_bf16(false, (A), false, (B), (short)0, (C), false, false)

// ---- Fused SwiGLU FFN on one 32-token block (256 threads = 8 waves) -------
// y[m, 0:H) = gate[m] * ( (silu(x W1^T + b1) * (x W3^T + b3)) W2^T + b2 )
// scattered to out rows lds_dst[m].  Two 16-row M-tiles share every B-fragment.
__device__ __forceinline__ void ffn_block(
    const float* __restrict__ x,
    const float* __restrict__ w1e, const float* __restrict__ b1e,
    const float* __restrict__ w3e, const float* __restrict__ b3e,
    const float* __restrict__ w2e, const float* __restrict__ b2e,
    float* __restrict__ outbase,
    __bf16* lds_x, __bf16* lds_act,
    const int* lds_tok, const int* lds_dst, const float* lds_gate) {
  const int tid  = threadIdx.x;
  const int lane = tid & 31;
  const int wave = tid >> 5;
  const int lm   = lane & 15;
  const int half = lane >> 4;

  // Gather MB token rows (fp32) -> bf16 LDS tile [MB][H]
  for (int i = tid; i < MB * (Hd / 4); i += 256) {
    int m  = i / (Hd / 4);
    int c4 = i % (Hd / 4);
    int t  = lds_tok[m];
    float4 v = make_float4(0.f, 0.f, 0.f, 0.f);
    if (t >= 0) v = ((const float4*)(x + (size_t)t * Hd))[c4];
    __bf16* d = lds_x + m * Hd + c4 * 4;
    d[0] = (__bf16)v.x; d[1] = (__bf16)v.y;
    d[2] = (__bf16)v.z; d[3] = (__bf16)v.w;
  }
  __syncthreads();

  // Stage 1: h1/h3 [MB, I], each wave owns 128 columns (8 N-tiles of 16)
  for (int nt = 0; nt < 8; ++nt) {
    const int n = wave * 128 + nt * 16 + lm;
    v8f acc1lo = {}, acc1hi = {}, acc3lo = {}, acc3hi = {};
    const float* w1r = w1e + (size_t)n * Hd;
    const float* w3r = w3e + (size_t)n * Hd;
#pragma unroll 2
    for (int kb = 0; kb < Hd; kb += 32) {
      v16bf a0 = load_a_lds(lds_x, Hd, lm, kb, half);
      v16bf a1 = load_a_lds(lds_x + 16 * Hd, Hd, lm, kb, half);
      v16bf b1 = load_b_f32(w1r, kb, half);
      acc1lo = WMMA_BF16(a0, b1, acc1lo);
      acc1hi = WMMA_BF16(a1, b1, acc1hi);
      v16bf b3 = load_b_f32(w3r, kb, half);
      acc3lo = WMMA_BF16(a0, b3, acc3lo);
      acc3hi = WMMA_BF16(a1, b3, acc3hi);
    }
    const float bb1 = b1e[n];
    const float bb3 = b3e[n];
#pragma unroll
    for (int r = 0; r < 8; ++r) {
      {
        float h1  = acc1lo[r] + bb1;
        float h3  = acc3lo[r] + bb3;
        float sig = 1.f / (1.f + __expf(-h1));
        lds_act[(r + 8 * half) * Id + n] = (__bf16)(h1 * sig * h3);
      }
      {
        float h1  = acc1hi[r] + bb1;
        float h3  = acc3hi[r] + bb3;
        float sig = 1.f / (1.f + __expf(-h1));
        lds_act[(16 + r + 8 * half) * Id + n] = (__bf16)(h1 * sig * h3);
      }
    }
  }
  __syncthreads();

  // Stage 2: y [MB, H] = act @ W2^T, each wave owns 64 columns (4 N-tiles)
  for (int nt = 0; nt < 4; ++nt) {
    const int n = wave * 64 + nt * 16 + lm;
    v8f acclo = {}, acchi = {};
    const float* w2r = w2e + (size_t)n * Id;
#pragma unroll 2
    for (int kb = 0; kb < Id; kb += 32) {
      v16bf a0 = load_a_lds(lds_act, Id, lm, kb, half);
      v16bf a1 = load_a_lds(lds_act + 16 * Id, Id, lm, kb, half);
      v16bf b  = load_b_f32(w2r, kb, half);
      acclo = WMMA_BF16(a0, b, acclo);
      acchi = WMMA_BF16(a1, b, acchi);
    }
    const float bb2 = b2e[n];
#pragma unroll
    for (int r = 0; r < 8; ++r) {
      int m0 = r + 8 * half;
      int m1 = 16 + m0;
      int d0 = lds_dst[m0];
      int d1 = lds_dst[m1];
      if (d0 >= 0) outbase[(size_t)d0 * Hd + n] = lds_gate[m0] * (acclo[r] + bb2);
      if (d1 >= 0) outbase[(size_t)d1 * Hd + n] = lds_gate[m1] * (acchi[r] + bb2);
    }
  }
}

// ---- Kernels --------------------------------------------------------------
__global__ void moe_zero(int* __restrict__ counts) {
  counts[threadIdx.x] = 0;
}

// One 64-thread block per token: logits, exact top-4, gates, scatter to lists.
__global__ void moe_router(const float* __restrict__ x,
                           const float* __restrict__ rw,
                           const float* __restrict__ rb,
                           const float* __restrict__ ebias,
                           int* __restrict__ counts,
                           int* __restrict__ list,
                           float* __restrict__ glist) {
  const int t = blockIdx.x;
  const int e = threadIdx.x;  // 0..63
  __shared__ float lg[Ed];    // original logits
  __shared__ float lgb[Ed];   // bias-adjusted (selection) logits

  const float4* xr = (const float4*)(x + (size_t)t * Hd);
  const float4* wr = (const float4*)(rw + (size_t)e * Hd);
  float s = 0.f;
  for (int h = 0; h < Hd / 4; ++h) {
    float4 a = xr[h], b = wr[h];
    s += a.x * b.x + a.y * b.y + a.z * b.z + a.w * b.w;
  }
  s += rb[e];
  lg[e]  = s;
  lgb[e] = s + ebias[e];
  __syncthreads();

  if (e == 0) {
    unsigned long long used = 0ull;
    int   idx[Kd];
    float sc[Kd];
    float ssum = 0.f;
    for (int k = 0; k < Kd; ++k) {
      float best = -3.4e38f;
      int   bi   = 0;
      for (int j = 0; j < Ed; ++j) {
        if (!((used >> j) & 1ull) && lgb[j] > best) { best = lgb[j]; bi = j; }
      }
      used |= (1ull << bi);
      idx[k] = bi;
      float sg = 1.f / (1.f + __expf(-lg[bi]));  // gate from ORIGINAL logit
      sc[k] = sg;
      ssum += sg;
    }
    ssum = fmaxf(ssum, 1e-12f);
    for (int k = 0; k < Kd; ++k) {
      int pos = atomicAdd(&counts[idx[k]], 1);
      list[idx[k] * Td + pos]  = t * Kd + k;  // (token, slot) packed
      glist[idx[k] * Td + pos] = sc[k] / ssum;
    }
  }
}

// Shared expert: dense over all tokens; one block per MB-token tile.
__global__ void moe_shared(const float* __restrict__ x,
                           const float* __restrict__ sw1, const float* __restrict__ sb1,
                           const float* __restrict__ sw3, const float* __restrict__ sb3,
                           const float* __restrict__ sw2, const float* __restrict__ sb2,
                           float* __restrict__ sbuf) {
  __shared__ __bf16 lds_x[MB * Hd];
  __shared__ __bf16 lds_act[MB * Id];
  __shared__ int    lds_tok[MB];
  __shared__ int    lds_dst[MB];
  __shared__ float  lds_gate[MB];

  if (threadIdx.x < MB) {
    int t = blockIdx.x * MB + threadIdx.x;
    lds_tok[threadIdx.x]  = t;
    lds_dst[threadIdx.x]  = t;
    lds_gate[threadIdx.x] = 1.f;
  }
  __syncthreads();
  ffn_block(x, sw1, sb1, sw3, sb3, sw2, sb2, sbuf,
            lds_x, lds_act, lds_tok, lds_dst, lds_gate);
}

// Routed experts: grid (E, 4); WG (e, j) handles M-blocks j, j+4, ... of expert e.
__global__ void moe_expert(const float* __restrict__ x,
                           const int* __restrict__ counts,
                           const int* __restrict__ list,
                           const float* __restrict__ glist,
                           const float* __restrict__ w1, const float* __restrict__ b1,
                           const float* __restrict__ w3, const float* __restrict__ b3,
                           const float* __restrict__ w2, const float* __restrict__ b2,
                           float* __restrict__ buf) {
  __shared__ __bf16 lds_x[MB * Hd];
  __shared__ __bf16 lds_act[MB * Id];
  __shared__ int    lds_tok[MB];
  __shared__ int    lds_dst[MB];
  __shared__ float  lds_gate[MB];

  const int e   = blockIdx.x;
  const int n_e = counts[e];
  const float* w1e = w1 + (size_t)e * Id * Hd;
  const float* b1e = b1 + (size_t)e * Id;
  const float* w3e = w3 + (size_t)e * Id * Hd;
  const float* b3e = b3 + (size_t)e * Id;
  const float* w2e = w2 + (size_t)e * Hd * Id;
  const float* b2e = b2 + (size_t)e * Hd;

  for (int blk = blockIdx.y; blk * MB < n_e; blk += gridDim.y) {
    __syncthreads();  // protect LDS reuse across iterations
    if (threadIdx.x < MB) {
      int i = blk * MB + threadIdx.x;
      if (i < n_e) {
        int ent = list[e * Td + i];
        lds_tok[threadIdx.x]  = ent >> 2;  // token
        lds_dst[threadIdx.x]  = ent;       // token*K + slot -> private row in buf
        lds_gate[threadIdx.x] = glist[e * Td + i];
      } else {
        lds_tok[threadIdx.x]  = -1;
        lds_dst[threadIdx.x]  = -1;
        lds_gate[threadIdx.x] = 0.f;
      }
    }
    __syncthreads();
    ffn_block(x, w1e, b1e, w3e, b3e, w2e, b2e, buf,
              lds_x, lds_act, lds_tok, lds_dst, lds_gate);
  }
}

// out = shared + sum over the K private slot rows of each token.
__global__ void moe_combine(const float* __restrict__ sbuf,
                            const float* __restrict__ buf,
                            float* __restrict__ out) {
  int i4 = blockIdx.x * blockDim.x + threadIdx.x;  // over T*H/4 float4s
  int el = i4 * 4;
  int t  = el / Hd;
  int h  = el % Hd;
  float4 s = ((const float4*)sbuf)[i4];
#pragma unroll
  for (int k = 0; k < Kd; ++k) {
    float4 v = *(const float4*)(buf + ((size_t)(t * Kd + k)) * Hd + h);
    s.x += v.x; s.y += v.y; s.z += v.z; s.w += v.w;
  }
  ((float4*)out)[i4] = s;
}

// ---- Host launch ----------------------------------------------------------
extern "C" void kernel_launch(void* const* d_in, const int* in_sizes, int n_in,
                              void* d_out, int out_size, void* d_ws, size_t ws_size,
                              hipStream_t stream) {
  (void)in_sizes; (void)n_in; (void)out_size; (void)ws_size;
  const float* x     = (const float*)d_in[0];
  const float* rw    = (const float*)d_in[1];
  const float* rb    = (const float*)d_in[2];
  const float* ebias = (const float*)d_in[3];
  const float* sw1   = (const float*)d_in[4];
  const float* sb1   = (const float*)d_in[5];
  const float* sw2   = (const float*)d_in[6];
  const float* sb2   = (const float*)d_in[7];
  const float* sw3   = (const float*)d_in[8];
  const float* sb3   = (const float*)d_in[9];
  const float* w1    = (const float*)d_in[10];
  const float* b1    = (const float*)d_in[11];
  const float* w2    = (const float*)d_in[12];
  const float* b2    = (const float*)d_in[13];
  const float* w3    = (const float*)d_in[14];
  const float* b3    = (const float*)d_in[15];
  float* out = (float*)d_out;

  char* p = (char*)d_ws;
  int*   counts = (int*)p;                          p += 4096;
  int*   list   = (int*)p;                          p += (size_t)Ed * Td * sizeof(int);
  float* glist  = (float*)p;                        p += (size_t)Ed * Td * sizeof(float);
  float* buf    = (float*)p;                        p += (size_t)Td * Kd * Hd * sizeof(float);
  float* sbuf   = (float*)p;

  moe_zero<<<1, Ed, 0, stream>>>(counts);
  moe_router<<<Td, Ed, 0, stream>>>(x, rw, rb, ebias, counts, list, glist);
  moe_shared<<<Td / MB, 256, 0, stream>>>(x, sw1, sb1, sw3, sb3, sw2, sb2, sbuf);
  moe_expert<<<dim3(Ed, 4), 256, 0, stream>>>(x, counts, list, glist,
                                              w1, b1, w3, b3, w2, b2, buf);
  moe_combine<<<(Td * Hd / 4) / 256, 256, 0, stream>>>(sbuf, buf, out);
}
